// Model_7773890806494
// MI455X (gfx1250) — matrix operand order repaired and checked by reading, not compiled, not run on hardware
//
#include <hip/hip_runtime.h>
#include <math.h>

namespace {

constexpr int kB = 32;        // batch
constexpr int kN = 4096;      // nodes
constexpr int kC = 64;        // channels (in == out)
constexpr int kM = 64;        // hyperedges
constexpr int kDeg = 256;     // incidences per edge
constexpr int kE = kM * kDeg; // 16384 incidences
constexpr float kNegSlope = 0.2f;
constexpr float kMargin = 4.2f;

typedef __attribute__((ext_vector_type(16))) _Float16 v16h;
typedef __attribute__((ext_vector_type(8)))  _Float16 v8h;
typedef __attribute__((ext_vector_type(8)))  float    v8f;
typedef __attribute__((ext_vector_type(4)))  unsigned int u32x4;
typedef __attribute__((ext_vector_type(8)))  int      i32x8;
typedef __attribute__((ext_vector_type(4)))  int      i32x4;

__device__ __forceinline__ void atomic_max_f32(float* addr, float val) {
  unsigned int* ua = reinterpret_cast<unsigned int*>(addr);
  unsigned int cur = *ua;
  while (__uint_as_float(cur) < val) {
    unsigned int assumed = cur;
    cur = atomicCAS(ua, assumed, __float_as_uint(val));
    if (cur == assumed) break;
  }
}

// pack 16 f32 (4x float4, contiguous) into a v16h fragment
__device__ __forceinline__ v16h cvt16(float4 f0, float4 f1, float4 f2, float4 f3) {
  v16h r;
  r[0]  = (_Float16)f0.x; r[1]  = (_Float16)f0.y; r[2]  = (_Float16)f0.z; r[3]  = (_Float16)f0.w;
  r[4]  = (_Float16)f1.x; r[5]  = (_Float16)f1.y; r[6]  = (_Float16)f1.z; r[7]  = (_Float16)f1.w;
  r[8]  = (_Float16)f2.x; r[9]  = (_Float16)f2.y; r[10] = (_Float16)f2.z; r[11] = (_Float16)f2.w;
  r[12] = (_Float16)f3.x; r[13] = (_Float16)f3.y; r[14] = (_Float16)f3.z; r[15] = (_Float16)f3.w;
  return r;
}

// load 16 contiguous f16 from LDS as two b128 reads
__device__ __forceinline__ v16h ld16h(const _Float16* p) {
  v8h lo = *(const v8h*)p;
  v8h hi = *(const v8h*)(p + 8);
  return __builtin_shufflevector(lo, hi, 0, 1, 2, 3, 4, 5, 6, 7, 8, 9, 10, 11, 12, 13, 14, 15);
}

// ---------- init node_max to -inf ----------
__global__ void k_neg_inf(float* __restrict__ p, int n) {
  int i = blockIdx.x * blockDim.x + threadIdx.x;
  if (i < n) p[i] = -INFINITY;
}

// ---------- xw = x @ W via WMMA, stored transposed as x1[N][B][C] ----------
// x: [B,N,C] f32 row-major; w: [C,C] f32. One wave: 16 flat rows x 64 cols.
// W is DMA'd into LDS by the Tensor Data Mover (one descriptor per block),
// transposed/converted to f16 in LDS, then each B fragment is two ds_load_b128.
// Output goes through an LDS store-transpose so stores are global_store_b128.
__global__ void k_xw_wmma(const float* __restrict__ x, const float* __restrict__ w,
                          float* __restrict__ x1) {
  __shared__ float    wraw[kC * kC];      // 16 KB raw f32 W (TDM destination)
  __shared__ _Float16 wt[kC * kC];        // 8 KB: wt[c*64 + k] = (f16) w[k*64 + c]
  __shared__ float    tile[8][16 * kC];   // 32 KB: per-wave store-transpose buffer

#if __has_builtin(__builtin_amdgcn_tensor_load_to_lds)
  if (threadIdx.x < 32) {  // wave 0 issues the tensor DMA (EXEC ignored by TDM)
    const unsigned long long ga = (unsigned long long)(const void*)w;
    const unsigned lds = (unsigned)(unsigned long long)(const void*)wraw;  // aperture: addr[31:0] = LDS offset
    u32x4 g0;
    g0[0] = 1u;                                     // count=1, user mode, no gather
    g0[1] = lds;                                    // lds_addr (bytes)
    g0[2] = (unsigned)(ga & 0xffffffffu);           // global_addr[31:0]
    g0[3] = (unsigned)((ga >> 32) & 0x01ffffffu)    // global_addr[56:32]
            | (2u << 30);                           // type = 2 ("image")
    i32x8 g1;
    g1[0] = (2 << 16);   // data_size = 4 bytes; wg_mask = 0; no pad/iterate
    g1[1] = (64 << 16);  // tensor_dim0 = 64 (low half)
    g1[2] = (64 << 16);  // tensor_dim1 = 64
    g1[3] = (64 << 16);  // tile_dim0 = 64
    g1[4] = 64;          // tile_dim1 = 64 (tile_dim2 = 0)
    g1[5] = 64;          // tensor_dim0_stride = 64 elements
    g1[6] = 0;           // tensor_dim1_stride unused (2D)
    g1[7] = 0;
    const i32x4 gz4 = {0, 0, 0, 0};
    const i32x8 gz8 = {0, 0, 0, 0, 0, 0, 0, 0};
    __builtin_amdgcn_tensor_load_to_lds(g0, g1, gz4, gz4, gz8, 0);
    __builtin_amdgcn_s_wait_tensorcnt(0);
  }
  __syncthreads();
  {
    const int t = threadIdx.x;  // 256 threads x 16 elements: transpose + cvt in LDS
#pragma unroll
    for (int u = 0; u < 16; ++u) {
      const int j = t * 16 + u;
      const int k = j >> 6, c = j & 63;
      wt[c * kC + k] = (_Float16)wraw[j];
    }
  }
#else
  {
    const int t = threadIdx.x;
#pragma unroll
    for (int u = 0; u < 16; ++u) {
      const int j = t * 16 + u;
      const int k = j >> 6, c = j & 63;
      wt[c * kC + k] = (_Float16)w[j];
    }
  }
#endif
  __syncthreads();

  const int lane = threadIdx.x & 31;
  const int wid  = threadIdx.x >> 5;
  const int wave = blockIdx.x * (blockDim.x >> 5) + wid;
  const int rowBase = wave * 16;            // flat row index in [0, B*N)
  const int half = lane >> 4;               // 0: lanes 0-15, 1: lanes 16-31
  const int l16  = lane & 15;

  // A fragments (16x32 f16): lane holds row rowBase+l16; fragment element s
  // maps to K = 16*half + s, so each fragment is one 16-float chunk (b128 loads).
  const float4* __restrict__ x4 = (const float4*)(x + (size_t)(rowBase + l16) * kC);
  const v16h a0 = cvt16(x4[half * 4],      x4[half * 4 + 1],
                        x4[half * 4 + 2],  x4[half * 4 + 3]);    // K = 0..31
  const v16h a1 = cvt16(x4[half * 4 + 8],  x4[half * 4 + 9],
                        x4[half * 4 + 10], x4[half * 4 + 11]);   // K = 32..63

  // B fragments from transposed LDS weight: 16 contiguous f16 per fragment.
  v16h bf[2][4];
#pragma unroll
  for (int kt = 0; kt < 2; ++kt) {
#pragma unroll
    for (int nt = 0; nt < 4; ++nt) {
      bf[kt][nt] = ld16h(&wt[(nt * 16 + l16) * kC + kt * 32 + half * 16]);
    }
  }

  v8f acc[4] = {};
#pragma unroll
  for (int nt = 0; nt < 4; ++nt) {
    acc[nt] = __builtin_amdgcn_wmma_f32_16x16x32_f16(false, a0, false, bf[0][nt],
                                                     (short)0, acc[nt], false, false);
    acc[nt] = __builtin_amdgcn_wmma_f32_16x16x32_f16(false, a1, false, bf[1][nt],
                                                     (short)0, acc[nt], false, false);
  }

  // D layout: VGPR i -> M = i + 8*half, N = lane%16. Stage in LDS, then store
  // full rows as float4 so the 33.5 MB output stream uses global_store_b128.
#pragma unroll
  for (int nt = 0; nt < 4; ++nt) {
#pragma unroll
    for (int i = 0; i < 8; ++i) {
      tile[wid][(i + 8 * half) * kC + nt * 16 + l16] = acc[nt][i];
    }
  }
  __syncthreads();
#pragma unroll
  for (int j = 0; j < 8; ++j) {
    const int s = lane * 8 + j;          // 256 float4 slots per wave tile
    const int row = s >> 4;              // 0..15
    const int c4  = s & 15;
    const int r = rowBase + row;         // flat row = b*kN + n
    const int b = r >> 12;
    const int n = r & (kN - 1);
    const float4 v = ((const float4*)&tile[wid][row * kC])[c4];
    ((float4*)(x1 + ((size_t)n * kB + b) * kC))[c4] = v;
  }
}

// ---------- edge_sums[m][b][c] = sum over incidences of x1[node] ----------
__global__ void k_edge_sum(const float* __restrict__ x1, const int* __restrict__ node_ids,
                           float* __restrict__ es) {
  __shared__ int nid[kDeg];
  const int m = blockIdx.x;
  nid[threadIdx.x] = node_ids[m * kDeg + threadIdx.x];
  __syncthreads();
  const int q = blockIdx.y * blockDim.x + threadIdx.x;  // 0..511 float4 slot
  float4 acc = {0.f, 0.f, 0.f, 0.f};
  for (int d = 0; d < kDeg; ++d) {
    if (d + 4 < kDeg)
      __builtin_prefetch((const float4*)(x1 + (size_t)nid[d + 4] * (kB * kC)) + q, 0, 0);
    const float4 v = ((const float4*)(x1 + (size_t)nid[d] * (kB * kC)))[q];
    acc.x += v.x; acc.y += v.y; acc.z += v.z; acc.w += v.w;
  }
  ((float4*)(es + (size_t)m * (kB * kC)))[q] = acc;
}

// ---------- per-(m,b) L2 norm of edge_sums + global sum (for mean(x_i - x_j)) ----------
__global__ void k_norm_sum(const float* __restrict__ es, float* __restrict__ nrm,
                           float* __restrict__ scal) {
  const int idx = blockIdx.x * blockDim.x + threadIdx.x;  // 0..2047 = m*32+b
  const float4* __restrict__ r4 = (const float4*)(es + (size_t)idx * kC);
  float ss = 0.f, sm = 0.f;
#pragma unroll
  for (int c4 = 0; c4 < kC / 4; ++c4) {
    const float4 v = r4[c4];
    ss += v.x * v.x + v.y * v.y + v.z * v.z + v.w * v.w;
    sm += v.x + v.y + v.z + v.w;
  }
  nrm[idx] = sqrtf(ss);
  atomicAdd(&scal[1], sm);
}

// ---------- pairwise loss via WMMA Gram matrix ----------
// inner[k,m,b] = <es_k, es_m>; dist^2 = |es_k|^2 + |es_m|^2 - 2*inner.
__global__ void k_inner_wmma(const float* __restrict__ es, const float* __restrict__ nrm,
                             float* __restrict__ lossKM) {
  const int tile = blockIdx.x;       // 0..15
  const int kt = tile >> 2, mt = tile & 3;
  const int b = blockIdx.y;          // 0..31
  const int lane = threadIdx.x & 31;
  const int half = lane >> 4, l16 = lane & 15;

  const float4* __restrict__ ar =
      (const float4*)(es + ((size_t)(kt * 16 + l16) * kB + b) * kC);
  const v16h a0 = cvt16(ar[half * 2],      ar[half * 2 + 1],
                        ar[half * 2 + 4],  ar[half * 2 + 5]);
  const v16h a1 = cvt16(ar[half * 2 + 8],  ar[half * 2 + 9],
                        ar[half * 2 + 12], ar[half * 2 + 13]);

  const float4* __restrict__ br =
      (const float4*)(es + ((size_t)(mt * 16 + l16) * kB + b) * kC);
  const v16h b0 = cvt16(br[half * 2],     br[half * 2 + 1],
                        br[half * 2 + 4], br[half * 2 + 5]);         // c = 0..31
  const v16h b1 = cvt16(br[8 + half * 2],     br[8 + half * 2 + 1],
                        br[8 + half * 2 + 4], br[8 + half * 2 + 5]); // c = 32..63

  v8f acc = {};
  acc = __builtin_amdgcn_wmma_f32_16x16x32_f16(false, a0, false, b0,
                                               (short)0, acc, false, false);
  acc = __builtin_amdgcn_wmma_f32_16x16x32_f16(false, a1, false, b1,
                                               (short)0, acc, false, false);

#pragma unroll
  for (int i = 0; i < 8; ++i) {
    const int kr = kt * 16 + i + 8 * half;
    const int mc = mt * 16 + l16;
    const float inner = acc[i];
    const float nk = nrm[kr * kB + b], nm = nrm[mc * kB + b];
    const float cosv = inner / (nk * nm);
    const float dist = sqrtf(fmaxf(nk * nk + nm * nm - 2.f * inner, 0.f));
    const float li = cosv * dist + (1.f - cosv) * fmaxf(kMargin - dist, 0.f);
    atomicAdd(&lossKM[kr * kM + mc], li);
  }
}

// ---------- reduce lossKM -> scal[0] ----------
__global__ void k_loss_reduce(const float* __restrict__ lossKM, float* __restrict__ scal) {
  const int i = blockIdx.x * blockDim.x + threadIdx.x;  // < M*M
  float v = fabsf(lossKM[i] * (1.f / kB)) * (1.f / ((kM + 1) * (kM + 1)));
#pragma unroll
  for (int off = 16; off > 0; off >>= 1) v += __shfl_down(v, off, 32);
  if ((threadIdx.x & 31) == 0) atomicAdd(&scal[0], v);
}

// ---------- attention scores + per-node max + node degree ----------
__global__ void k_scores(const float* __restrict__ x1, const float* __restrict__ es,
                         const int* __restrict__ node_ids, const float* __restrict__ att,
                         float* __restrict__ sA, float* __restrict__ nmax,
                         float* __restrict__ Dn) {
  const int e = blockIdx.x;
  const int b = threadIdx.x;             // 0..31
  const int node = node_ids[e];
  const int m = e >> 8;                  // edge_ids = repeat(arange(M), DEG)
  const float4* __restrict__ xi = (const float4*)(x1 + ((size_t)node * kB + b) * kC);
  const float4* __restrict__ xj = (const float4*)(es + ((size_t)m * kB + b) * kC);
  const float4* __restrict__ ai = (const float4*)att;        // att[0..63]
  const float4* __restrict__ aj = (const float4*)(att + kC); // att[64..127]
  float sc = 0.f;
#pragma unroll
  for (int c4 = 0; c4 < kC / 4; ++c4) {
    const float4 u = xi[c4], v = xj[c4], p = ai[c4], q = aj[c4];
    sc += u.x * p.x + u.y * p.y + u.z * p.z + u.w * p.w;
    sc += v.x * q.x + v.y * q.y + v.z * q.z + v.w * q.w;
  }
  sc = (sc > 0.f) ? sc : kNegSlope * sc;
  sA[(size_t)e * kB + b] = sc;
  atomic_max_f32(&nmax[node * kB + b], sc);
  if (b == 0) atomicAdd(&Dn[node], 1.0f);
}

// ---------- exp(score - max) and per-node sum ----------
__global__ void k_expsum(const int* __restrict__ node_ids, float* __restrict__ sA,
                         const float* __restrict__ nmax, float* __restrict__ nsum) {
  const int idx = blockIdx.x * blockDim.x + threadIdx.x;  // < E*B
  const int e = idx >> 5, b = idx & 31;
  const int node = node_ids[e];
  const float v = expf(sA[idx] - nmax[node * kB + b]);
  sA[idx] = v;
  atomicAdd(&nsum[node * kB + b], v);
}

// ---------- normalize: alpha = e / segment_sum ----------
__global__ void k_alpha(const int* __restrict__ node_ids, float* __restrict__ sA,
                        const float* __restrict__ nsum) {
  const int idx = blockIdx.x * blockDim.x + threadIdx.x;
  const int e = idx >> 5, b = idx & 31;
  sA[idx] = sA[idx] / nsum[node_ids[e] * kB + b];
}

// ---------- stage 1: nodes -> hyperedges, out_e = (1/DEG) * sum alpha * x_i ----------
__global__ void k_stage1(const float* __restrict__ x1, const int* __restrict__ node_ids,
                         const float* __restrict__ sA, float* __restrict__ oute) {
  __shared__ int nid[kDeg];
  const int m = blockIdx.x;
  nid[threadIdx.x] = node_ids[m * kDeg + threadIdx.x];
  __syncthreads();
  const int q = blockIdx.y * blockDim.x + threadIdx.x;  // 0..511 float4 slot
  const int b = q >> 4, c4 = q & 15;
  float4 acc = {0.f, 0.f, 0.f, 0.f};
  for (int d = 0; d < kDeg; ++d) {
    const int e = m * kDeg + d;
    const float al = sA[(size_t)e * kB + b];
    const float4 v = ((const float4*)(x1 + ((size_t)nid[d] * kB + b) * kC))[c4];
    acc.x += al * v.x; acc.y += al * v.y; acc.z += al * v.z; acc.w += al * v.w;
  }
  const float s = 1.0f / kDeg;
  acc.x *= s; acc.y *= s; acc.z *= s; acc.w *= s;
  ((float4*)(oute + (size_t)m * (kB * kC)))[q] = acc;
}

// ---------- stage 2: hyperedges -> nodes, scatter into out[B,N,C] ----------
__global__ void k_stage2(const int* __restrict__ node_ids, const float* __restrict__ sA,
                         const float* __restrict__ oute, const float* __restrict__ Dn,
                         float* __restrict__ out) {
  const int e = blockIdx.x;
  const int node = node_ids[e];
  const int m = e >> 8;
  const float Dv = Dn[node];
  for (int q = threadIdx.x; q < (kB * kC) / 4; q += blockDim.x) {
    const int b = q >> 4, c4 = q & 15;
    const float al = Dv * sA[(size_t)e * kB + b];
    const float4 v = ((const float4*)(oute + (size_t)m * (kB * kC)))[q];
    float* dst = out + ((size_t)b * kN + node) * kC + c4 * 4;
    atomicAdd(dst + 0, al * v.x);
    atomicAdd(dst + 1, al * v.y);
    atomicAdd(dst + 2, al * v.z);
    atomicAdd(dst + 3, al * v.w);
  }
}

// ---------- final scalar: |mean(x_i - x_j)| + loss_hyper ----------
__global__ void k_final(const float* __restrict__ scal, float* __restrict__ out, int last) {
  // sum(x_i - x_j) = (1 - DEG) * sum(edge_sums)  (algebraic identity)
  const float denom = (float)((size_t)kE * kB * kC);
  out[last] = fabsf(scal[1]) * ((float)(kDeg - 1) / denom) + scal[0];
}

}  // namespace

extern "C" void kernel_launch(void* const* d_in, const int* in_sizes, int n_in,
                              void* d_out, int out_size, void* d_ws, size_t ws_size,
                              hipStream_t stream) {
  const float* x   = (const float*)d_in[0];   // [B,N,C]
  const float* w   = (const float*)d_in[1];   // [C,C]
  const float* att = (const float*)d_in[2];   // [1,1,2C] -> 128 floats
  const int* hidx  = (const int*)d_in[3];     // [2,E]: row0 node_ids, row1 edge_ids
  const int* node_ids = hidx;                 // edge_ids implicit (sorted repeat)

  float* out = (float*)d_out;                 // [B,N,C] flat + 1 scalar

  // ---- workspace partition (floats) ----
  float* ws     = (float*)d_ws;
  float* x1     = ws;                            // N*B*C = 8,388,608
  float* es     = x1 + (size_t)kN * kB * kC;     // M*B*C = 131,072
  float* nrm    = es + (size_t)kM * kB * kC;     // M*B   = 2,048
  float* sA     = nrm + kM * kB;                 // E*B   = 524,288
  float* nmax   = sA + (size_t)kE * kB;          // N*B   = 131,072
  float* nsum   = nmax + kN * kB;                // N*B   = 131,072
  float* oute   = nsum + kN * kB;                // M*B*C = 131,072
  float* Dn     = oute + (size_t)kM * kB * kC;   // N     = 4,096
  float* lossKM = Dn + kN;                       // M*M   = 4,096
  float* scal   = lossKM + kM * kM;              // [0]=loss, [1]=sum(edge_sums)

  // ---- capture-safe initialization ----
  (void)hipMemsetAsync(out, 0, (size_t)out_size * sizeof(float), stream);
  (void)hipMemsetAsync(nsum, 0, (size_t)kN * kB * sizeof(float), stream);
  (void)hipMemsetAsync(Dn, 0, (size_t)kN * sizeof(float), stream);
  (void)hipMemsetAsync(lossKM, 0, (size_t)kM * kM * sizeof(float), stream);
  (void)hipMemsetAsync(scal, 0, 2 * sizeof(float), stream);
  k_neg_inf<<<(kN * kB + 255) / 256, 256, 0, stream>>>(nmax, kN * kB);

  // ---- 1) GEMM (WMMA + TDM weight staging) -> transposed x1[N][B][C] ----
  k_xw_wmma<<<(kB * kN) / (16 * 8), 256, 0, stream>>>(x, w, x1);

  // ---- 2) per-edge segment sums ----
  k_edge_sum<<<dim3(kM, 2), 256, 0, stream>>>(x1, node_ids, es);

  // ---- 3) norms + global edge-sum reduction ----
  k_norm_sum<<<(kM * kB) / 256, 256, 0, stream>>>(es, nrm, scal);

  // ---- 4) pairwise hyperedge loss: WMMA Gram matrix + reduce ----
  k_inner_wmma<<<dim3(16, kB), 32, 0, stream>>>(es, nrm, lossKM);
  k_loss_reduce<<<(kM * kM) / 256, 256, 0, stream>>>(lossKM, scal);

  // ---- 5) attention scores + segment max + degrees ----
  k_scores<<<kE, 32, 0, stream>>>(x1, es, node_ids, att, sA, nmax, Dn);

  // ---- 6) segment softmax (exp-sum then normalize) ----
  k_expsum<<<(kE * kB) / 256, 256, 0, stream>>>(node_ids, sA, nmax, nsum);
  k_alpha<<<(kE * kB) / 256, 256, 0, stream>>>(node_ids, sA, nsum);

  // ---- 7) two-stage propagation ----
  k_stage1<<<dim3(kM, 2), 256, 0, stream>>>(x1, node_ids, sA, oute);
  k_stage2<<<kE, 256, 0, stream>>>(node_ids, sA, oute, Dn, out);

  // ---- 8) final scalar ----
  k_final<<<1, 1, 0, stream>>>(scal, out, out_size - 1);
}
